// AttnBlock_3942779977824
// MI455X (gfx1250) — compile-verified, matrix operand order checked by reading, not run on hardware
//
#include <hip/hip_runtime.h>

// ---------------------------------------------------------------------------
// AttnBlock for MI455X (gfx1250): GroupNorm -> QKV 1x1 -> attention -> proj
// GEMMs via V_WMMA_F32_16X16X32_F16; tiles staged with
// GLOBAL_LOAD_ASYNC_TO_LDS_B128 (ASYNCcnt) into double-buffered LDS.
// ---------------------------------------------------------------------------

typedef __attribute__((ext_vector_type(16))) _Float16 v16h;
typedef __attribute__((ext_vector_type(8)))  _Float16 h8;
typedef __attribute__((ext_vector_type(2)))  _Float16 h2;
typedef __attribute__((ext_vector_type(8)))  float    v8f;

#define C_DIM 512
#define N_DIM 4096
#define B_DIM 2
#define NGRP  32
#define CPG   16           // channels per group
#define GELEM (CPG * N_DIM)

// ------------------------- CDNA5 async-copy helpers ------------------------
// Per-lane async copy of 16 bytes global -> LDS, tracked by ASYNCcnt.
// vdst VGPR carries the 32-bit LDS byte address (low 32 bits of flat shared
// pointer, per ISA 10.2 aperture mapping).
__device__ __forceinline__ void async_copy_b128(void* lds_ptr, const void* gptr) {
    unsigned lds_off = (unsigned)(unsigned long long)lds_ptr;
    asm volatile("global_load_async_to_lds_b128 %0, %1, off"
                 :: "v"(lds_off), "v"((unsigned long long)gptr)
                 : "memory");
}
__device__ __forceinline__ void wait_async0() {
    asm volatile("s_wait_asynccnt 0x0" ::: "memory");
}

// ------------------------- block reductions --------------------------------
__device__ __forceinline__ float block_reduce_sum(float v, float* buf) {
    int tid = threadIdx.x;
    buf[tid] = v; __syncthreads();
    #pragma unroll
    for (int s = 128; s > 0; s >>= 1) {
        if (tid < s) buf[tid] += buf[tid + s];
        __syncthreads();
    }
    float r = buf[0]; __syncthreads();
    return r;
}
__device__ __forceinline__ float block_reduce_max(float v, float* buf) {
    int tid = threadIdx.x;
    buf[tid] = v; __syncthreads();
    #pragma unroll
    for (int s = 128; s > 0; s >>= 1) {
        if (tid < s) buf[tid] = fmaxf(buf[tid], buf[tid + s]);
        __syncthreads();
    }
    float r = buf[0]; __syncthreads();
    return r;
}

// ------------------------- GroupNorm ---------------------------------------
__global__ __launch_bounds__(256) void gn_stats_kernel(
    const float* __restrict__ x, float* __restrict__ stats) {
    __shared__ float buf[256];
    const float* p = x + (size_t)blockIdx.x * GELEM;
    float s = 0.f, s2 = 0.f;
    for (int i = threadIdx.x; i < GELEM; i += 256) {
        float v = p[i];
        s += v; s2 += v * v;
    }
    s  = block_reduce_sum(s,  buf);
    s2 = block_reduce_sum(s2, buf);
    if (threadIdx.x == 0) {
        float mu  = s  * (1.0f / GELEM);
        float var = s2 * (1.0f / GELEM) - mu * mu;
        stats[blockIdx.x * 2 + 0] = mu;
        stats[blockIdx.x * 2 + 1] = rsqrtf(var + 1e-6f);
    }
}

// normalize + affine, write h^T as f16 [B][N][C] (k-contiguous for GEMM B)
__global__ __launch_bounds__(256) void gn_apply_kernel(
    const float* __restrict__ x, const float* __restrict__ stats,
    const float* __restrict__ gw, const float* __restrict__ gb,
    _Float16* __restrict__ hT) {
    int bc = blockIdx.x;               // b*C + c
    int b  = bc >> 9;
    int c  = bc & (C_DIM - 1);
    int g  = c >> 4;
    float mu = stats[(b * NGRP + g) * 2 + 0];
    float rs = stats[(b * NGRP + g) * 2 + 1];
    float w  = gw[c], bb = gb[c];
    const float* xr = x + (size_t)bc * N_DIM;
    _Float16* dst = hT + (size_t)b * N_DIM * C_DIM + c;
    for (int n = threadIdx.x; n < N_DIM; n += 256) {
        float v = (xr[n] - mu) * rs * w + bb;
        dst[(size_t)n * C_DIM] = (_Float16)v;
    }
}

// ------------------------- f32 -> f16 convert ------------------------------
__global__ __launch_bounds__(256) void cvt_f16_kernel(
    const float* __restrict__ src, _Float16* __restrict__ dst, int n) {
    for (int i = blockIdx.x * 256 + threadIdx.x; i < n; i += gridDim.x * 256)
        dst[i] = (_Float16)src[i];
}

// ------------------------- WMMA fragment load ------------------------------
// LDS tile is row-major [r][32] halves. Per ISA 16-bit A 16x32 layout:
// lane<16 -> K base 0, lane>=16 -> K base 8; elems 0..7 = K base..base+7,
// elems 8..15 = K base+16..base+23. B (stored transposed [n][k]) symmetric.
__device__ __forceinline__ v16h load_frag32(const _Float16* lds, int row, int kbase) {
    const h8 lo = *(const h8*)(lds + row * 32 + kbase);
    const h8 hi = *(const h8*)(lds + row * 32 + kbase + 16);
    return __builtin_shufflevector(lo, hi, 0,1,2,3,4,5,6,7,8,9,10,11,12,13,14,15);
}

// ------------------------- generic WMMA GEMM -------------------------------
// D = alpha * A(f16 [M][K]) x B(f16) + bias(+residual)
// OUT_MODE: 0 = f16 [M][N] (+bias)      (V, attn-out)
//           1 = f16 [N][M] (+bias)      (Q, K^T)
//           2 = f32 [M][N], scaled      (scores)
//           3 = f32 [M][N] + bias + residual (final proj)
// B_KN: true  -> global B is [K][N] (ldb = N pitch), transposed into LDS
//       false -> global B is [N][K] (pre-transposed), async direct copy
template <int OUT_MODE, bool B_KN>
__global__ __launch_bounds__(256) void gemm_wmma_kernel(
    const _Float16* __restrict__ A, int lda,
    const _Float16* __restrict__ Bm, int ldb,
    void* __restrict__ Out, int ldo,
    const float* __restrict__ bias,
    const float* __restrict__ residual,
    float alpha, int M, int N, int K) {
    (void)M; (void)N;
    __shared__ _Float16 lA[2][128 * 32];
    __shared__ _Float16 lB[2][128 * 32];

    const int tid   = threadIdx.x;
    const int lane  = tid & 31;
    const int wave  = tid >> 5;     // 0..7
    const int waveM = wave & 3;     // 4 -> 32 rows each
    const int waveN = wave >> 2;    // 2 -> 64 cols each
    const int m0g   = blockIdx.x * 128;
    const int n0g   = blockIdx.y * 128;

    v8f acc[2][4];
    #pragma unroll
    for (int i = 0; i < 2; ++i)
        #pragma unroll
        for (int j = 0; j < 4; ++j)
            acc[i][j] = (v8f){0.f,0.f,0.f,0.f,0.f,0.f,0.f,0.f};

    const int rl    = lane & 15;
    const int kbase = (lane < 16) ? 0 : 8;

    // ---- tile staging (async for direct paths; packed DS for transpose) ----
    auto stage = [&](int bufi, int k0) {
        // A tile [128][32]: 512 16-byte chunks, 2 per thread, memory -> LDS
        #pragma unroll
        for (int it = 0; it < 2; ++it) {
            int cc = tid + it * 256;
            int m  = cc >> 2;
            int kq = (cc & 3) * 8;
            async_copy_b128(&lA[bufi][m * 32 + kq],
                            A + (size_t)(m0g + m) * lda + k0 + kq);
        }
        if (!B_KN) {
            #pragma unroll
            for (int it = 0; it < 2; ++it) {
                int cc = tid + it * 256;
                int n  = cc >> 2;
                int kq = (cc & 3) * 8;
                async_copy_b128(&lB[bufi][n * 32 + kq],
                                Bm + (size_t)(n0g + n) * ldb + k0 + kq);
            }
        } else {
            // transpose fill: pair k rows, packed 2xf16 ds_store_b32
            int k2 = (tid & 15) * 2;          // even k
            int nq = (tid >> 4) * 8;          // 0..120
            h8 v0 = *(const h8*)(Bm + (size_t)(k0 + k2)     * ldb + n0g + nq);
            h8 v1 = *(const h8*)(Bm + (size_t)(k0 + k2 + 1) * ldb + n0g + nq);
            #pragma unroll
            for (int e = 0; e < 8; ++e) {
                h2 t = {v0[e], v1[e]};
                *(h2*)(&lB[bufi][(nq + e) * 32 + k2]) = t;
            }
        }
    };

    int buf = 0;
    stage(buf, 0);
    wait_async0();
    __syncthreads();

    for (int k0 = 0; k0 < K; k0 += 32) {
        int nbuf = buf ^ 1;
        if (k0 + 32 < K) stage(nbuf, k0 + 32);   // overlap with compute

        // ---- compute: one K=32 WMMA step from current buffer
        v16h af[2], bf[4];
        #pragma unroll
        for (int sm = 0; sm < 2; ++sm)
            af[sm] = load_frag32(lA[buf], waveM * 32 + sm * 16 + rl, kbase);
        #pragma unroll
        for (int sn = 0; sn < 4; ++sn)
            bf[sn] = load_frag32(lB[buf], waveN * 64 + sn * 16 + rl, kbase);
        #pragma unroll
        for (int sm = 0; sm < 2; ++sm)
            #pragma unroll
            for (int sn = 0; sn < 4; ++sn)
                acc[sm][sn] = __builtin_amdgcn_wmma_f32_16x16x32_f16(
                    false, af[sm], false, bf[sn],
                    (short)0, acc[sm][sn], false, false);

        wait_async0();        // next-buffer async writes landed
        __syncthreads();      // + all waves done reading current buffer
        buf = nbuf;
    }

    // ---- epilogue: lane holds col n = rl, rows mb..mb+7
    const int mhalf = (lane < 16) ? 0 : 8;
    #pragma unroll
    for (int sm = 0; sm < 2; ++sm) {
        #pragma unroll
        for (int sn = 0; sn < 4; ++sn) {
            int mb = m0g + waveM * 32 + sm * 16 + mhalf;
            int n  = n0g + waveN * 64 + sn * 16 + rl;
            v8f a = acc[sm][sn];
            if (OUT_MODE == 0) {
                _Float16* O = (_Float16*)Out;
                #pragma unroll
                for (int r = 0; r < 8; ++r) {
                    float v = a[r] * alpha + (bias ? bias[mb + r] : 0.0f);
                    O[(size_t)(mb + r) * ldo + n] = (_Float16)v;
                }
            } else if (OUT_MODE == 1) {
                _Float16* O = (_Float16*)Out;
                h8 pack;
                #pragma unroll
                for (int r = 0; r < 8; ++r)
                    pack[r] = (_Float16)(a[r] * alpha + (bias ? bias[mb + r] : 0.0f));
                *(h8*)(O + (size_t)n * ldo + mb) = pack;   // contiguous over m
            } else if (OUT_MODE == 2) {
                float* O = (float*)Out;
                #pragma unroll
                for (int r = 0; r < 8; ++r)
                    O[(size_t)(mb + r) * ldo + n] = a[r] * alpha;
            } else {
                float* O = (float*)Out;
                #pragma unroll
                for (int r = 0; r < 8; ++r) {
                    size_t idx = (size_t)(mb + r) * ldo + n;
                    O[idx] = residual[idx] + a[r] * alpha + bias[mb + r];
                }
            }
        }
    }
}

// ------------------------- row softmax (in-place f32 -> f16) ---------------
// S: f32 [N][N]. Writes normalized f16 over the head of each row:
// P element (i,j) lives at half index i*2N + j  (pitch 2N halves).
__global__ __launch_bounds__(256) void softmax_kernel(float* __restrict__ S) {
    __shared__ float buf[256];
    const int row = blockIdx.x;
    float* r = S + (size_t)row * N_DIM;
    float vals[16];
    float lmax = -3.4e38f;
    #pragma unroll
    for (int i = 0; i < 16; ++i) {
        vals[i] = r[i * 256 + threadIdx.x];
        lmax = fmaxf(lmax, vals[i]);
    }
    float m = block_reduce_max(lmax, buf);
    float lsum = 0.f;
    #pragma unroll
    for (int i = 0; i < 16; ++i) {
        vals[i] = __expf(vals[i] - m);
        lsum += vals[i];
    }
    float s = block_reduce_sum(lsum, buf);   // barrier: all reads done
    float inv = 1.0f / s;
    _Float16* P = (_Float16*)S + (size_t)row * (2 * N_DIM);
    #pragma unroll
    for (int i = 0; i < 16; ++i)
        P[i * 256 + threadIdx.x] = (_Float16)(vals[i] * inv);
}

// ---------------------------------------------------------------------------
extern "C" void kernel_launch(void* const* d_in, const int* in_sizes, int n_in,
                              void* d_out, int out_size, void* d_ws, size_t ws_size,
                              hipStream_t stream) {
    (void)in_sizes; (void)n_in; (void)out_size; (void)ws_size;
    const float* x      = (const float*)d_in[0];
    const float* norm_w = (const float*)d_in[1];
    const float* norm_b = (const float*)d_in[2];
    const float* wq = (const float*)d_in[3];
    const float* bq = (const float*)d_in[4];
    const float* wk = (const float*)d_in[5];
    const float* bk = (const float*)d_in[6];
    const float* wv = (const float*)d_in[7];
    const float* bv = (const float*)d_in[8];
    const float* wp = (const float*)d_in[9];
    const float* bp = (const float*)d_in[10];
    float* out = (float*)d_out;

    // ---- carve workspace
    char* w = (char*)d_ws;
    auto carve = [&](size_t bytes) -> char* {
        char* p = w;
        w += (bytes + 255) & ~(size_t)255;
        return p;
    };
    float*     stats = (float*)carve(B_DIM * NGRP * 2 * sizeof(float));
    _Float16*  hT    = (_Float16*)carve((size_t)B_DIM * N_DIM * C_DIM * 2); // [b][n][c]
    _Float16*  w16q  = (_Float16*)carve((size_t)C_DIM * C_DIM * 2);
    _Float16*  w16k  = (_Float16*)carve((size_t)C_DIM * C_DIM * 2);
    _Float16*  w16v  = (_Float16*)carve((size_t)C_DIM * C_DIM * 2);
    _Float16*  w16p  = (_Float16*)carve((size_t)C_DIM * C_DIM * 2);
    _Float16*  Q     = (_Float16*)carve((size_t)B_DIM * N_DIM * C_DIM * 2); // [b][n][c]
    _Float16*  Kt    = (_Float16*)carve((size_t)B_DIM * N_DIM * C_DIM * 2); // [b][n][c]
    _Float16*  V     = (_Float16*)carve((size_t)B_DIM * C_DIM * N_DIM * 2); // [b][c][n]
    _Float16*  AO    = (_Float16*)carve((size_t)B_DIM * C_DIM * N_DIM * 2); // [b][c][n]
    float*     S     = (float*)carve((size_t)N_DIM * N_DIM * sizeof(float)); // per-batch

    const float scale = 0.044194173824159216f;  // 512^-0.5

    // 1) GroupNorm
    gn_stats_kernel<<<B_DIM * NGRP, 256, 0, stream>>>(x, stats);
    gn_apply_kernel<<<B_DIM * C_DIM, 256, 0, stream>>>(x, stats, norm_w, norm_b, hT);

    // 2) weights to f16
    cvt_f16_kernel<<<256, 256, 0, stream>>>(wq, w16q, C_DIM * C_DIM);
    cvt_f16_kernel<<<256, 256, 0, stream>>>(wk, w16k, C_DIM * C_DIM);
    cvt_f16_kernel<<<256, 256, 0, stream>>>(wv, w16v, C_DIM * C_DIM);
    cvt_f16_kernel<<<256, 256, 0, stream>>>(wp, w16p, C_DIM * C_DIM);

    dim3 gQKV(C_DIM / 128, N_DIM / 128);   // 4 x 32
    dim3 gS(N_DIM / 128, N_DIM / 128);     // 32 x 32

    for (int b = 0; b < B_DIM; ++b) {
        const _Float16* hTb = hT + (size_t)b * N_DIM * C_DIM;
        _Float16* Qb  = Q  + (size_t)b * N_DIM * C_DIM;
        _Float16* Ktb = Kt + (size_t)b * N_DIM * C_DIM;
        _Float16* Vb  = V  + (size_t)b * C_DIM * N_DIM;
        // 3) QKV projections (M=512, N=4096, K=512), B operand = hT "nk" direct
        gemm_wmma_kernel<1, false><<<gQKV, 256, 0, stream>>>(
            w16q, C_DIM, hTb, C_DIM, Qb,  C_DIM, bq, nullptr, 1.0f,
            C_DIM, N_DIM, C_DIM);
        gemm_wmma_kernel<1, false><<<gQKV, 256, 0, stream>>>(
            w16k, C_DIM, hTb, C_DIM, Ktb, C_DIM, bk, nullptr, 1.0f,
            C_DIM, N_DIM, C_DIM);
        gemm_wmma_kernel<0, false><<<gQKV, 256, 0, stream>>>(
            w16v, C_DIM, hTb, C_DIM, Vb,  N_DIM, bv, nullptr, 1.0f,
            C_DIM, N_DIM, C_DIM);
    }

    for (int b = 0; b < B_DIM; ++b) {
        const _Float16* Qb  = Q  + (size_t)b * N_DIM * C_DIM;
        const _Float16* Ktb = Kt + (size_t)b * N_DIM * C_DIM;
        const _Float16* Vb  = V  + (size_t)b * C_DIM * N_DIM;
        _Float16* AOb = AO + (size_t)b * C_DIM * N_DIM;
        // 4) scores: S[i][j] = scale * Q[i][:] . Kt[j][:]   (M=N=4096, K=512)
        gemm_wmma_kernel<2, false><<<gS, 256, 0, stream>>>(
            Qb, C_DIM, Ktb, C_DIM, S, N_DIM, nullptr, nullptr, scale,
            N_DIM, N_DIM, C_DIM);
        // 5) softmax rows, normalized f16 written in place (pitch 2N halves)
        softmax_kernel<<<N_DIM, 256, 0, stream>>>(S);
        // 6) AO[c][j] = sum_i V[c][i] * P[i][j]  (M=512, N=4096, K=4096)
        gemm_wmma_kernel<0, true><<<gQKV, 256, 0, stream>>>(
            Vb, N_DIM, (const _Float16*)S, 2 * N_DIM, AOb, N_DIM,
            nullptr, nullptr, 1.0f, C_DIM, N_DIM, N_DIM);
        // 7) out = x + Wp @ AO + bp   (M=512, N=4096, K=512)
        gemm_wmma_kernel<3, true><<<gQKV, 256, 0, stream>>>(
            w16p, C_DIM, AOb, N_DIM, out + (size_t)b * C_DIM * N_DIM, N_DIM,
            bp, x + (size_t)b * C_DIM * N_DIM, 1.0f, C_DIM, N_DIM, C_DIM);
    }
}
